// FeaturePropagationModule_87050397155551
// MI455X (gfx1250) — matrix operand order repaired and checked by reading, not compiled, not run on hardware
//
#include <hip/hip_runtime.h>
#include <hip/hip_fp16.h>

// Problem constants (match reference)
#define B_    8
#define N1_   16384
#define N2_   2048
#define C1_   128
#define C2_   256
#define CIN_  384
#define H1_   256
#define H2_   128
#define M_    (B_ * N1_)        // 131072 total points
#define EPS_W  1e-5f
#define EPS_BN 1e-5f

typedef __attribute__((ext_vector_type(16))) _Float16 v16h;
typedef __attribute__((ext_vector_type(8)))  _Float16 v8h;
typedef __attribute__((ext_vector_type(8)))  float    v8f;
typedef __attribute__((ext_vector_type(4)))  int      v4i;

union V16H { v16h v; v8h h[2]; };

// ---------------------------------------------------------------------------
// 16-byte global -> LDS copy via the gfx1250 async-DMA path
// (global_load_async_to_lds_b128, tracked by ASYNCcnt). The builtin takes
// int4* pointers (global src, LDS dst), an immediate offset and cpol.
// ---------------------------------------------------------------------------
#if __has_builtin(__builtin_amdgcn_global_load_async_to_lds_b128)
#define HAVE_ASYNC_LDS 1
#else
#define HAVE_ASYNC_LDS 0
#endif

__device__ __forceinline__ void async_copy16(const _Float16* g, _Float16* l) {
#if HAVE_ASYNC_LDS
  __builtin_amdgcn_global_load_async_to_lds_b128(
      (__attribute__((address_space(1))) v4i*)(g),
      (__attribute__((address_space(3))) v4i*)(l), 0, 0);
#else
  *(v8h*)l = *(const v8h*)g;
#endif
}

__device__ __forceinline__ void wait_async_copies() {
#if HAVE_ASYNC_LDS
#if __has_builtin(__builtin_amdgcn_s_wait_asynccnt)
  __builtin_amdgcn_s_wait_asynccnt(0);
#else
  asm volatile("s_wait_asynccnt 0" ::: "memory");
#endif
#endif
}

// ---------------------------------------------------------------------------
// Prep: convert w1/w2 to f16 (weights stay L2-resident), zero BN stat buffers.
// Must run every launch (graph replays don't re-poison, atomics accumulate).
// ---------------------------------------------------------------------------
__global__ __launch_bounds__(256) void prep_kernel(
    const float* __restrict__ w1, const float* __restrict__ w2,
    _Float16* __restrict__ w1h, _Float16* __restrict__ w2h,
    float* __restrict__ stats /* 2*(H1+H2) floats: sum1,sq1,sum2,sq2 */) {
  int i = blockIdx.x * 256 + threadIdx.x;
  if (i < H1_ * CIN_) w1h[i] = (_Float16)w1[i];
  if (i < H2_ * H1_)  w2h[i] = (_Float16)w2[i];
  if (i < 2 * (H1_ + H2_)) stats[i] = 0.0f;
}

// ---------------------------------------------------------------------------
// 3-NN + inverse-distance interpolation + concat -> x[m][k] f16, k=0..383
// One thread per query point; points2 (24 KB) staged in LDS per block.
// ---------------------------------------------------------------------------
__global__ __launch_bounds__(256) void interp_concat(
    const float* __restrict__ p1, const float* __restrict__ f1,
    const float* __restrict__ p2, const float* __restrict__ f2,
    _Float16* __restrict__ x) {
  __shared__ float px[N2_], py[N2_], pz[N2_];
  const int b  = blockIdx.x >> 6;            // 64 blocks per batch (N1/256)
  const int n0 = (blockIdx.x & 63) << 8;
  const float* p2b = p2 + (size_t)b * 3 * N2_;
  for (int j = threadIdx.x; j < N2_; j += 256) {
    px[j] = p2b[j];
    py[j] = p2b[N2_ + j];
    pz[j] = p2b[2 * N2_ + j];
  }
  __syncthreads();

  const int n = n0 + threadIdx.x;
  const float* p1b = p1 + (size_t)b * 3 * N1_;
  const float X = p1b[n], Y = p1b[N1_ + n], Z = p1b[2 * N1_ + n];

  float d0 = 3.4e38f, d1 = 3.4e38f, d2 = 3.4e38f;
  int   i0 = 0, i1 = 0, i2 = 0;
  for (int j = 0; j < N2_; ++j) {
    float dx = X - px[j], dy = Y - py[j], dz = Z - pz[j];
    float d = dx * dx + dy * dy + dz * dz;
    if (d < d2) {
      if (d < d1) {
        d2 = d1; i2 = i1;
        if (d < d0) { d1 = d0; i1 = i0; d0 = d; i0 = j; }
        else        { d1 = d;  i1 = j; }
      } else { d2 = d; i2 = j; }
    }
  }
  float w0 = 1.0f / (d0 + EPS_W);
  float w1 = 1.0f / (d1 + EPS_W);
  float w2 = 1.0f / (d2 + EPS_W);
  float inv = 1.0f / (w0 + w1 + w2);
  w0 *= inv; w1 *= inv; w2 *= inv;

  const size_t m = (size_t)b * N1_ + n;
  _Float16* xr = x + m * CIN_;
  const float* f1b = f1 + (size_t)b * C1_ * N1_ + n;
#pragma unroll 4
  for (int c = 0; c < C1_; ++c) xr[c] = (_Float16)f1b[(size_t)c * N1_];
  const float* f2b = f2 + (size_t)b * C2_ * N2_;
#pragma unroll 4
  for (int c = 0; c < C2_; ++c) {
    const float* fc = f2b + (size_t)c * N2_;
    float v = w0 * fc[i0] + w1 * fc[i1] + w2 * fc[i2];
    xr[C1_ + c] = (_Float16)v;
  }
}

// ---------------------------------------------------------------------------
// GEMM1: y1[o][m] = sum_k w1[o][k]*x[m][k] + b1[o]    (o<256, k<384)
// 1024-thread blocks (32 waves). blockIdx.y selects a 128-channel half of w1,
// which is staged once into LDS (pitch 392 f16 -> conflict-free ds_load_b128).
// Each wave then owns 8 o-tiles x 1 m-tile; x is streamed from global once
// per channel-half. BN1 stats fused via shfl butterfly + LDS + global atomics.
// ---------------------------------------------------------------------------
__global__ __launch_bounds__(1024) void gemm1_wmma(
    const _Float16* __restrict__ w1h, const float* __restrict__ bias,
    const _Float16* __restrict__ x, float* __restrict__ y1,
    float* __restrict__ gsum, float* __restrict__ gsq) {
  constexpr int ROWS  = H1_ / 2;    // 128 output channels per block
  constexpr int PITCH = CIN_ + 8;   // 392 f16 = 784B: lanes hit distinct banks
  __shared__ _Float16 wlds[ROWS * PITCH];   // ~98 KB
  __shared__ float lsum[ROWS], lsq[ROWS];

  const int tid = threadIdx.x;
  if (tid < ROWS) { lsum[tid] = 0.0f; lsq[tid] = 0.0f; }

  const int ohalf = blockIdx.y;
  const _Float16* wsrc = w1h + (size_t)ohalf * ROWS * CIN_;
  // Stage 128x384 f16 via async 16B copies (6 per thread)
  for (int i = tid; i < ROWS * (CIN_ / 8); i += 1024) {
    int row = i / (CIN_ / 8), c = i % (CIN_ / 8);
    async_copy16(wsrc + (size_t)row * CIN_ + c * 8, &wlds[row * PITCH + c * 8]);
  }
  wait_async_copies();
  __syncthreads();

  const int wave = tid >> 5;
  const int lane = tid & 31;
  const int half = lane >> 4;       // K-half select: k = 16*half + s in A and B (this code's order)
  const int l16  = lane & 15;
  const int mt   = blockIdx.x * 32 + wave;
  const int mcol = mt * 16 + l16;

  v8f acc[8];
#pragma unroll
  for (int ot = 0; ot < 8; ++ot)
#pragma unroll
    for (int r = 0; r < 8; ++r) acc[ot][r] = 0.0f;

  const _Float16* xrow = x + (size_t)mcol * CIN_;
  for (int kb = 0; kb < CIN_; kb += 32) {
    // B fragment: 16 contiguous f16 (K = kb+half*16 .. +15) for column mcol
    V16H bf;
    bf.v = *(const v16h*)(xrow + kb + half * 16);
#pragma unroll
    for (int ot = 0; ot < 8; ++ot) {
      // A fragment from LDS: row o = ot*16 + l16, two 8-elem K chunks
      const _Float16* ap = &wlds[(ot * 16 + l16) * PITCH + kb + half * 16];
      V16H af;
      af.h[0] = *(const v8h*)ap;
      af.h[1] = *(const v8h*)(ap + 8);
      acc[ot] = __builtin_amdgcn_wmma_f32_16x16x32_f16(
          false, af.v, false, bf.v, (short)0, acc[ot], false, false);
    }
  }

  // Epilogue: bias, store y1, BN stats
#pragma unroll
  for (int ot = 0; ot < 8; ++ot) {
#pragma unroll
    for (int r = 0; r < 8; ++r) {
      const int ol = ot * 16 + r + half * 8;        // 0..127 within half
      const int og = ohalf * ROWS + ol;             // global channel
      float v = acc[ot][r] + bias[og];
      y1[(size_t)og * M_ + mcol] = v;
      float s = v, q = v * v;
      s += __shfl_xor(s, 1);  q += __shfl_xor(q, 1);
      s += __shfl_xor(s, 2);  q += __shfl_xor(q, 2);
      s += __shfl_xor(s, 4);  q += __shfl_xor(q, 4);
      s += __shfl_xor(s, 8);  q += __shfl_xor(q, 8);
      if (l16 == 0) { atomicAdd(&lsum[ol], s); atomicAdd(&lsq[ol], q); }
    }
  }
  __syncthreads();
  if (tid < ROWS) {
    atomicAdd(&gsum[ohalf * ROWS + tid], lsum[tid]);
    atomicAdd(&gsq[ohalf * ROWS + tid], lsq[tid]);
  }
}

// ---------------------------------------------------------------------------
// BN coefficients: scale = g*rsqrt(var+eps), shift = beta - mu*scale
// ---------------------------------------------------------------------------
__global__ void bn_coeffs(const float* __restrict__ sum, const float* __restrict__ sq,
                          const float* __restrict__ g, const float* __restrict__ beta,
                          float* __restrict__ scale, float* __restrict__ shift, int C) {
  int c = blockIdx.x * blockDim.x + threadIdx.x;
  if (c >= C) return;
  const float invM = 1.0f / (float)M_;
  float mu  = sum[c] * invM;
  float var = sq[c] * invM - mu * mu;
  float sc  = g[c] * rsqrtf(var + EPS_BN);
  scale[c] = sc;
  shift[c] = beta[c] - mu * sc;
}

// ---------------------------------------------------------------------------
// GEMM2: all of w2 (128x256 f16) LDS-resident (pitch 264 -> conflict-free).
// Fused BN1+ReLU on the B-operand load (y1 f32 -> f16); 8 o-tiles per wave;
// writes pre-BN2 result directly into d_out in final (B,H2,N1) layout and
// accumulates BN2 stats.
// ---------------------------------------------------------------------------
__global__ __launch_bounds__(1024) void gemm2_wmma(
    const _Float16* __restrict__ w2h, const float* __restrict__ bias,
    const float* __restrict__ y1,
    const float* __restrict__ scale1, const float* __restrict__ shift1,
    float* __restrict__ out, float* __restrict__ gsum, float* __restrict__ gsq) {
  constexpr int PITCH = H1_ + 8;    // 264 f16 = 528B: conflict-free rows
  __shared__ _Float16 w2lds[H2_ * PITCH];   // ~66 KB
  __shared__ float lsum[H2_], lsq[H2_];

  const int tid = threadIdx.x;
  if (tid < H2_) { lsum[tid] = 0.0f; lsq[tid] = 0.0f; }

  // Stage 128x256 f16 via async 16B copies (4 per thread)
  for (int i = tid; i < H2_ * (H1_ / 8); i += 1024) {
    int row = i / (H1_ / 8), c = i % (H1_ / 8);
    async_copy16(w2h + (size_t)row * H1_ + c * 8, &w2lds[row * PITCH + c * 8]);
  }
  wait_async_copies();
  __syncthreads();

  const int wave = tid >> 5;
  const int lane = tid & 31;
  const int half = lane >> 4;
  const int l16  = lane & 15;
  const int mt   = blockIdx.x * 32 + wave;
  const int mcol = mt * 16 + l16;

  v8f acc[8];
#pragma unroll
  for (int ot = 0; ot < 8; ++ot)
#pragma unroll
    for (int r = 0; r < 8; ++r) acc[ot][r] = 0.0f;

  for (int kb = 0; kb < H1_; kb += 32) {
    V16H bf;
#pragma unroll
    for (int e = 0; e < 16; ++e) {
      const int k = kb + half * 16 + e;
      float v = y1[(size_t)k * M_ + mcol];      // coalesced across l16 lanes
      v = v * scale1[k] + shift1[k];            // BN1
      v = fmaxf(v, 0.0f);                       // ReLU
      bf.v[e] = (_Float16)v;
    }
#pragma unroll
    for (int ot = 0; ot < 8; ++ot) {
      const _Float16* ap = &w2lds[(ot * 16 + l16) * PITCH + kb + half * 16];
      V16H af;
      af.h[0] = *(const v8h*)ap;
      af.h[1] = *(const v8h*)(ap + 8);
      acc[ot] = __builtin_amdgcn_wmma_f32_16x16x32_f16(
          false, af.v, false, bf.v, (short)0, acc[ot], false, false);
    }
  }

  const int b = mcol >> 14;            // mcol / N1
  const int n = mcol & (N1_ - 1);
  float* ob = out + (size_t)b * H2_ * N1_ + n;
#pragma unroll
  for (int ot = 0; ot < 8; ++ot) {
#pragma unroll
    for (int r = 0; r < 8; ++r) {
      const int o = ot * 16 + r + half * 8;
      float v = acc[ot][r] + bias[o];
      ob[(size_t)o * N1_] = v;                  // pre-BN2, final layout
      float s = v, q = v * v;
      s += __shfl_xor(s, 1);  q += __shfl_xor(q, 1);
      s += __shfl_xor(s, 2);  q += __shfl_xor(q, 2);
      s += __shfl_xor(s, 4);  q += __shfl_xor(q, 4);
      s += __shfl_xor(s, 8);  q += __shfl_xor(q, 8);
      if (l16 == 0) { atomicAdd(&lsum[o], s); atomicAdd(&lsq[o], q); }
    }
  }
  __syncthreads();
  if (tid < H2_) { atomicAdd(&gsum[tid], lsum[tid]); atomicAdd(&gsq[tid], lsq[tid]); }
}

// ---------------------------------------------------------------------------
// Elementwise BN2 + ReLU, in-place on d_out (layout B,H2,N1 -> o = (i>>14)&127)
// ---------------------------------------------------------------------------
__global__ __launch_bounds__(256) void bn2_apply(
    float* __restrict__ out, const float* __restrict__ scale2,
    const float* __restrict__ shift2) {
  size_t i = (size_t)blockIdx.x * 256 + threadIdx.x;
  int o = (int)((i >> 14) & (H2_ - 1));
  float v = out[i] * scale2[o] + shift2[o];
  out[i] = fmaxf(v, 0.0f);
}

// ---------------------------------------------------------------------------
extern "C" void kernel_launch(void* const* d_in, const int* in_sizes, int n_in,
                              void* d_out, int out_size, void* d_ws, size_t ws_size,
                              hipStream_t stream) {
  (void)in_sizes; (void)n_in; (void)out_size; (void)ws_size;
  const float* p1  = (const float*)d_in[0];
  const float* f1  = (const float*)d_in[1];
  const float* p2  = (const float*)d_in[2];
  const float* f2  = (const float*)d_in[3];
  const float* w1  = (const float*)d_in[4];
  const float* b1  = (const float*)d_in[5];
  const float* g1  = (const float*)d_in[6];
  const float* be1 = (const float*)d_in[7];
  const float* w2  = (const float*)d_in[8];
  const float* b2  = (const float*)d_in[9];
  const float* g2  = (const float*)d_in[10];
  const float* be2 = (const float*)d_in[11];
  float* out = (float*)d_out;

  // Workspace carve-up (256B aligned slices)
  char* ws = (char*)d_ws;
  size_t off = 0;
  auto take = [&](size_t bytes) -> char* {
    char* p = ws + off;
    off += (bytes + 255) & ~(size_t)255;
    return p;
  };
  _Float16* x    = (_Float16*)take((size_t)M_ * CIN_ * sizeof(_Float16)); // 100.7 MB
  _Float16* w1h  = (_Float16*)take((size_t)H1_ * CIN_ * sizeof(_Float16));
  _Float16* w2h  = (_Float16*)take((size_t)H2_ * H1_ * sizeof(_Float16));
  float*    y1   = (float*)take((size_t)H1_ * M_ * sizeof(float));        // 134.2 MB
  float*    sum1 = (float*)take(H1_ * sizeof(float));   // stats block is contiguous:
  float*    sq1  = (float*)take(H1_ * sizeof(float));   // sizes are 256B multiples
  float*    sum2 = (float*)take(H2_ * sizeof(float));
  float*    sq2  = (float*)take(H2_ * sizeof(float));
  float*    scale1 = (float*)take(H1_ * sizeof(float));
  float*    shift1 = (float*)take(H1_ * sizeof(float));
  float*    scale2 = (float*)take(H2_ * sizeof(float));
  float*    shift2 = (float*)take(H2_ * sizeof(float));

  // 1. weight convert + zero stats (sum1 is base of the contiguous stat block)
  prep_kernel<<<(H1_ * CIN_ + 255) / 256, 256, 0, stream>>>(w1, w2, w1h, w2h, sum1);
  // 2. 3-NN + interpolate + concat -> x f16
  interp_concat<<<B_ * (N1_ / 256), 256, 0, stream>>>(p1, f1, p2, f2, x);
  // 3. GEMM1 + BN1 stats (grid.y = channel half; 32 m-tiles per block)
  gemm1_wmma<<<dim3(M_ / (16 * 32), 2), 1024, 0, stream>>>(w1h, b1, x, y1, sum1, sq1);
  // 4. BN1 coefficients
  bn_coeffs<<<1, H1_, 0, stream>>>(sum1, sq1, g1, be1, scale1, shift1, H1_);
  // 5. GEMM2 (fused BN1+ReLU on load) + BN2 stats, writes pre-BN2 into out
  gemm2_wmma<<<M_ / (16 * 32), 1024, 0, stream>>>(w2h, b2, y1, scale1, shift1,
                                                  out, sum2, sq2);
  // 6. BN2 coefficients
  bn_coeffs<<<1, H2_, 0, stream>>>(sum2, sq2, g2, be2, scale2, shift2, H2_);
  // 7. BN2 + ReLU in place
  bn2_apply<<<(size_t)H2_ * M_ / 256, 256, 0, stream>>>(out, scale2, shift2);
}
